// TemporalAdaptiveGNNStack_3496103379112
// MI455X (gfx1250) — compile-verified, matrix operand order checked by reading, not run on hardware
//
#include <hip/hip_runtime.h>
#include <hip/hip_bf16.h>
#include <math.h>

// Problem constants (match the reference)
#define B_  8
#define V_  64
#define P_  256
#define D_  128
#define ND_ 16
#define H_  4
#define FF_ 512
#define DH_ (D_ / H_)   // 32

typedef __attribute__((ext_vector_type(16))) __bf16 v16bf;
typedef __attribute__((ext_vector_type(8)))  float  v8f;

// A lane's 16 bf16 fragment values = two contiguous 16B chunks:
//   u[0..7]  = K(half*8 .. half*8+7)
//   u[8..15] = K(16+half*8 .. 16+half*8+7)
union BfFrag {
  v16bf v;
  unsigned short u[16];
  uint4 q[2];
};
union U4x8 {
  uint4 q;
  unsigned short u[8];
};

__device__ __forceinline__ unsigned short f2bf(float f) {
  unsigned int x = __float_as_uint(f);
  unsigned int r = (x + 0x7fffu + ((x >> 16) & 1u)) >> 16;  // RNE
  return (unsigned short)r;
}
__device__ __forceinline__ float hredmax16(float v) {
  for (int o = 8; o; o >>= 1) v = fmaxf(v, __shfl_xor(v, o));
  return v;
}

// -------------------------------------------------------------------------
// fp32 -> bf16 convert
__global__ void cvt_bf16_kernel(const float* __restrict__ src,
                                unsigned short* __restrict__ dst, int n) {
  int i = blockIdx.x * 256 + threadIdx.x;
  if (i < n) dst[i] = f2bf(src[i]);
}

// -------------------------------------------------------------------------
// h = x + sinusoidal PE ; write fp32 + bf16
__global__ void pe_kernel(const float* __restrict__ x,
                          float* __restrict__ Xf,
                          unsigned short* __restrict__ Xb) {
  size_t tok = blockIdx.x;              // over B*V*P
  int d = threadIdx.x;                  // 0..127
  int p = (int)(tok % P_);
  int i2 = (d >> 1) << 1;
  float div = __expf((float)i2 * (-logf(10000.f) / (float)D_));
  float ang = (float)p * div;
  float pe = (d & 1) ? __cosf(ang) : __sinf(ang);
  size_t off = tok * D_ + d;
  float v = x[off] + pe;
  Xf[off] = v;
  Xb[off] = f2bf(v);
}

// -------------------------------------------------------------------------
// Generic bf16 WMMA GEMM: out = act(A[M,K] @ W[K,N] + bias)
// Block: 256 threads = 8 waves; block tile 128(M) x 32(N); wave tile 16x32.
// W panel staged in LDS transposed (column-major: [32][K]) so every B-frag
// is two ds_load_b128; A-frags are two global_load_b128.
#define GF_RELU 1
#define GF_ACC  2
#define GF_F32  4
#define GF_BF16 8
template <int FLAGS>
__global__ void gemm_bf16_kernel(const unsigned short* __restrict__ A,
                                 const unsigned short* __restrict__ W,
                                 const float* __restrict__ bias,
                                 float* __restrict__ outF,
                                 unsigned short* __restrict__ outB,
                                 int M, int N, int K) {
  extern __shared__ __align__(16) unsigned short sW[];  // [32][K]
  const int tid = threadIdx.x;
  const int wave = tid >> 5;
  const int lane = tid & 31;
  const int half = lane >> 4;
  const int l15 = lane & 15;
  const int n0 = blockIdx.y * 32;
  const int m0 = blockIdx.x * 128 + wave * 16;

  // coalesced global read, transposed LDS write
  for (int i = tid; i < K * 32; i += 256) {
    int kk = i >> 5, nn = i & 31;
    sW[nn * K + kk] = W[(size_t)kk * N + n0 + nn];
  }
  __syncthreads();

  v8f c0 = {0.f, 0.f, 0.f, 0.f, 0.f, 0.f, 0.f, 0.f};
  v8f c1 = {0.f, 0.f, 0.f, 0.f, 0.f, 0.f, 0.f, 0.f};
  const unsigned short* Arow = A + (size_t)(m0 + l15) * K + half * 8;
  const unsigned short* Wc0 = sW + (size_t)l15 * K + half * 8;
  const unsigned short* Wc1 = sW + (size_t)(16 + l15) * K + half * 8;
  for (int kk = 0; kk < K; kk += 32) {
    if (kk + 32 < K) __builtin_prefetch(Arow + kk + 32, 0, 3);
    BfFrag a, b0, b1;
    a.q[0]  = *(const uint4*)(Arow + kk);
    a.q[1]  = *(const uint4*)(Arow + kk + 16);
    b0.q[0] = *(const uint4*)(Wc0 + kk);
    b0.q[1] = *(const uint4*)(Wc0 + kk + 16);
    b1.q[0] = *(const uint4*)(Wc1 + kk);
    b1.q[1] = *(const uint4*)(Wc1 + kk + 16);
    c0 = __builtin_amdgcn_wmma_f32_16x16x32_bf16(false, a.v, false, b0.v,
                                                 (short)0, c0, false, false);
    c1 = __builtin_amdgcn_wmma_f32_16x16x32_bf16(false, a.v, false, b1.v,
                                                 (short)0, c1, false, false);
  }

  float bv0 = bias ? bias[n0 + l15] : 0.f;
  float bv1 = bias ? bias[n0 + 16 + l15] : 0.f;
#pragma unroll
  for (int i = 0; i < 8; ++i) {
    int m = m0 + half * 8 + i;
    float y0 = c0[i] + bv0;
    float y1 = c1[i] + bv1;
    if (FLAGS & GF_RELU) { y0 = fmaxf(y0, 0.f); y1 = fmaxf(y1, 0.f); }
    size_t off0 = (size_t)m * N + n0 + l15;
    size_t off1 = off0 + 16;
    if (FLAGS & GF_F32) {
      float z0 = y0, z1 = y1;
      if (FLAGS & GF_ACC) { z0 += outF[off0]; z1 += outF[off1]; }
      outF[off0] = z0;
      outF[off1] = z1;
    }
    if (FLAGS & GF_BF16) {
      outB[off0] = f2bf(y0);
      outB[off1] = f2bf(y1);
    }
  }
}

// -------------------------------------------------------------------------
// LayerNorm over D: out = LN(base + delta)*g + b ; writes fp32 + bf16
__global__ void ln_kernel(const float* __restrict__ base,
                          const float* __restrict__ delta,
                          const float* __restrict__ g,
                          const float* __restrict__ b,
                          float* __restrict__ outF,
                          unsigned short* __restrict__ outB) {
  __shared__ float red1[4];
  __shared__ float red2[4];
  size_t row = blockIdx.x;
  int t = threadIdx.x;  // 0..127
  size_t off = row * D_ + t;
  float v = base[off] + delta[off];
  float s = v;
  for (int o = 16; o; o >>= 1) s += __shfl_xor(s, o);
  if ((t & 31) == 0) red1[t >> 5] = s;
  __syncthreads();
  float mean = (red1[0] + red1[1] + red1[2] + red1[3]) * (1.f / D_);
  float dv = v - mean;
  float q = dv * dv;
  for (int o = 16; o; o >>= 1) q += __shfl_xor(q, o);
  if ((t & 31) == 0) red2[t >> 5] = q;
  __syncthreads();
  float var = (red2[0] + red2[1] + red2[2] + red2[3]) * (1.f / D_);
  float y = dv * rsqrtf(var + 1e-5f) * g[t] + b[t];
  outF[off] = y;
  outB[off] = f2bf(y);
}

// -------------------------------------------------------------------------
// Flash-style masked attention. grid = (B*V, H, P/128), block = 256.
// K block kept [key][dh] (dh-contiguous) for score B-frags;
// V block stored transposed [dh][key] (key-contiguous) for p@V B-frags.
// Row sums of the probability tile come from an extra WMMA against an
// all-ones B fragment (replaces 32 ds_bpermute per block).
__global__ void attention_kernel(const unsigned short* __restrict__ Q,
                                 const unsigned short* __restrict__ Kb,
                                 const unsigned short* __restrict__ Vb,
                                 unsigned short* __restrict__ O,
                                 const int* __restrict__ patch,
                                 const int* __restrict__ stay) {
  __shared__ __align__(16) unsigned short sK[32 * DH_];   // [key][dh]
  __shared__ __align__(16) unsigned short sVt[DH_ * 32];  // [dh][key]
  __shared__ int sMsk[32];
  __shared__ __align__(16) unsigned short sP[8][16 * 32]; // per-wave p (A layout)

  const int tid = threadIdx.x;
  const int wave = tid >> 5;
  const int lane = tid & 31;
  const int half = lane >> 4;
  const int l15 = lane & 15;
  const int bv = blockIdx.x;
  const int b = bv / V_;
  const int v = bv % V_;
  const int h = blockIdx.y;
  const int m0 = blockIdx.z * 128 + wave * 16;
  const float scale = 0.17677669529663687f;  // 1/sqrt(32)

  // all-ones bf16 B fragment (for row sums via WMMA)
  BfFrag ones;
#pragma unroll
  for (int i = 0; i < 16; ++i) ones.u[i] = 0x3F80;

  // q fragment (16 rows x dh=32), two b128 loads
  BfFrag qa;
  {
    const unsigned short* qrow =
        Q + ((size_t)bv * P_ + m0 + l15) * D_ + h * DH_ + half * 8;
    qa.q[0] = *(const uint4*)(qrow);
    qa.q[1] = *(const uint4*)(qrow + 16);
  }

  v8f o0 = {0.f, 0.f, 0.f, 0.f, 0.f, 0.f, 0.f, 0.f};
  v8f o1 = {0.f, 0.f, 0.f, 0.f, 0.f, 0.f, 0.f, 0.f};
  float mrow[8], lsum[8];
#pragma unroll
  for (int i = 0; i < 8; ++i) { mrow[i] = -1e30f; lsum[i] = 0.f; }

  for (int kb0 = 0; kb0 < P_; kb0 += 32) {
    if (tid < 128) {
      int key = tid >> 2, c = tid & 3;  // 32 keys x 4 chunks of 8
      size_t goff = ((size_t)bv * P_ + kb0 + key) * D_ + h * DH_ + c * 8;
      uint4 kq = *(const uint4*)(Kb + goff);
      U4x8 vu;
      vu.q = *(const uint4*)(Vb + goff);
      *(uint4*)(sK + key * 32 + c * 8) = kq;
#pragma unroll
      for (int e = 0; e < 8; ++e) sVt[(c * 8 + e) * 32 + key] = vu.u[e];
    }
    if (tid < 32) {
      int key = kb0 + tid;
      sMsk[tid] =
          (stay[b * P_ + key] * patch[((size_t)b * V_ + v) * P_ + key]) == 0;
    }
    __syncthreads();

    // two 16-key score tiles:  s = q @ K^T
    float sv[2][8];
#pragma unroll
    for (int t = 0; t < 2; ++t) {
      BfFrag kf;
      const unsigned short* krow = sK + (t * 16 + l15) * 32 + half * 8;
      kf.q[0] = *(const uint4*)(krow);
      kf.q[1] = *(const uint4*)(krow + 16);
      v8f s = {0.f, 0.f, 0.f, 0.f, 0.f, 0.f, 0.f, 0.f};
      s = __builtin_amdgcn_wmma_f32_16x16x32_bf16(false, qa.v, false, kf.v,
                                                  (short)0, s, false, false);
      int msk = sMsk[t * 16 + l15];
#pragma unroll
      for (int i = 0; i < 8; ++i) sv[t][i] = msk ? -1e9f : s[i] * scale;
    }

    // online softmax: max via intra-half shuffles, rescale, stash p (bf16)
#pragma unroll
    for (int i = 0; i < 8; ++i) {
      float rm = hredmax16(fmaxf(sv[0][i], sv[1][i]));
      float nm = fmaxf(mrow[i], rm);
      float al = __expf(mrow[i] - nm);
      float p0 = __expf(sv[0][i] - nm);
      float p1 = __expf(sv[1][i] - nm);
      lsum[i] *= al;
      mrow[i] = nm;
      o0[i] *= al;
      o1[i] *= al;
      int row = half * 8 + i;
      sP[wave][row * 32 + l15]      = f2bf(p0);
      sP[wave][row * 32 + 16 + l15] = f2bf(p1);
    }

    // p fragments (C->A layout via per-wave LDS), V fragments
    BfFrag pa, vb0, vb1;
    {
      const unsigned short* prow = sP[wave] + l15 * 32 + half * 8;
      pa.q[0] = *(const uint4*)(prow);
      pa.q[1] = *(const uint4*)(prow + 16);
      const unsigned short* v0 = sVt + l15 * 32 + half * 8;
      vb0.q[0] = *(const uint4*)(v0);
      vb0.q[1] = *(const uint4*)(v0 + 16);
      const unsigned short* v1 = sVt + (16 + l15) * 32 + half * 8;
      vb1.q[0] = *(const uint4*)(v1);
      vb1.q[1] = *(const uint4*)(v1 + 16);
    }
    // row sums: p @ ones  (every column of the result equals the row sum)
    v8f rs = {0.f, 0.f, 0.f, 0.f, 0.f, 0.f, 0.f, 0.f};
    rs = __builtin_amdgcn_wmma_f32_16x16x32_bf16(false, pa.v, false, ones.v,
                                                 (short)0, rs, false, false);
    o0 = __builtin_amdgcn_wmma_f32_16x16x32_bf16(false, pa.v, false, vb0.v,
                                                 (short)0, o0, false, false);
    o1 = __builtin_amdgcn_wmma_f32_16x16x32_bf16(false, pa.v, false, vb1.v,
                                                 (short)0, o1, false, false);
#pragma unroll
    for (int i = 0; i < 8; ++i) lsum[i] += rs[i];
    __syncthreads();
  }

#pragma unroll
  for (int i = 0; i < 8; ++i) {
    float inv = 1.f / lsum[i];
    size_t base = ((size_t)bv * P_ + m0 + half * 8 + i) * D_ + h * DH_;
    O[base + l15]      = f2bf(o0[i] * inv);
    O[base + 16 + l15] = f2bf(o1[i] * inv);
  }
}

// -------------------------------------------------------------------------
// gvc1[v] = nodevec1[v,:] . g1W[D:D+ND]; gvc2[v] = nodevec2[:,v] . g2W[D:D+ND]
__global__ void gateconst_kernel(const float* __restrict__ nv1,
                                 const float* __restrict__ nv2,
                                 const float* __restrict__ g1W,
                                 const float* __restrict__ g2W,
                                 float* __restrict__ gvc1,
                                 float* __restrict__ gvc2) {
  int v = threadIdx.x;
  if (v >= V_) return;
  float s1 = 0.f, s2 = 0.f;
  for (int n = 0; n < ND_; ++n) {
    s1 += nv1[v * ND_ + n] * g1W[D_ + n];
    s2 += nv2[n * V_ + v] * g2W[D_ + n];
  }
  gvc1[v] = s1;
  gvc2[v] = s2;
}

// -------------------------------------------------------------------------
// Pack [lin1W | lin2W | gate1W(:D) | gate2W(:D) | 0...] into a 128x64 bf16
// panel + packed bias so the whole gating math runs as one WMMA GEMM.
__global__ void packgate_kernel(const float* __restrict__ lin1W,
                                const float* __restrict__ lin2W,
                                const float* __restrict__ g1W,
                                const float* __restrict__ g2W,
                                const float* __restrict__ lin1b,
                                const float* __restrict__ lin2b,
                                const float* __restrict__ g1b,
                                const float* __restrict__ g2b,
                                unsigned short* __restrict__ Wp,
                                float* __restrict__ biasp) {
  int i = blockIdx.x * 256 + threadIdx.x;  // over 128*64
  if (i < D_ * 64) {
    int d = i >> 6, c = i & 63;
    float w = 0.f;
    if (c < 16)       w = lin1W[d * ND_ + c];
    else if (c < 32)  w = lin2W[d * ND_ + (c - 16)];
    else if (c == 32) w = g1W[d];
    else if (c == 33) w = g2W[d];
    Wp[(size_t)d * 64 + c] = f2bf(w);
  }
  if (i < 64) {
    float bb = 0.f;
    if (i < 16)       bb = lin1b[i];
    else if (i < 32)  bb = lin2b[i - 16];
    else if (i == 32) bb = g1b[0];
    else if (i == 33) bb = g2b[0];
    biasp[i] = bb;
  }
}

// -------------------------------------------------------------------------
// Finish gating from the packed-GEMM result Tg[tok, 0..63]:
//   cols 0..15 = xe@lin1W+b, 16..31 = xe@lin2W+b, 32 = gate1 dot, 33 = gate2.
// One thread per (token, n). grid = B*V*P*16/256.
__global__ void gatefin_kernel(const float* __restrict__ Tg,
                               const float* __restrict__ nv1,
                               const float* __restrict__ nv2,
                               const float* __restrict__ gvc1,
                               const float* __restrict__ gvc2,
                               float* __restrict__ nv1o,
                               float* __restrict__ nv2o) {
  size_t idx = (size_t)blockIdx.x * 256 + threadIdx.x;
  size_t tok = idx >> 4;
  int n = (int)(idx & 15);
  int p = (int)(tok % P_);
  size_t bvi = tok / P_;
  int v = (int)(bvi % V_);
  int b = (int)(bvi / V_);
  const float* tg = Tg + tok * 64;
  float g1 = fmaxf(tanhf(tg[32] + gvc1[v]), 0.f);
  float g2 = fmaxf(tanhf(tg[33] + gvc2[v]), 0.f);
  nv1o[(((size_t)b * P_ + p) * V_ + v) * ND_ + n] = nv1[v * ND_ + n] + g1 * tg[n];
  nv2o[(((size_t)b * P_ + p) * ND_ + n) * V_ + v] = nv2[n * V_ + v] + g2 * tg[16 + n];
}

// -------------------------------------------------------------------------
// Adaptive adjacency: per (b,p) 64x64 = relu(nv1 @ nv2) row-softmax.
// Stored transposed bf16: adpT[bp, w, v]. grid = B*P, block = 256.
__global__ void adj_kernel(const float* __restrict__ nv1o,
                           const float* __restrict__ nv2o,
                           unsigned short* __restrict__ adpT) {
  __shared__ __align__(16) unsigned short sA[64 * 32];   // [v][k], K padded
  __shared__ __align__(16) unsigned short sBt[64 * 32];  // [w][k], K padded
  __shared__ float sS[64 * 66];
  const int tid = threadIdx.x;
  size_t bp = blockIdx.x;
  for (int i = tid; i < 64 * 32; i += 256) {
    int r = i >> 5, kk = i & 31;
    sA[i]  = (kk < ND_) ? f2bf(nv1o[(bp * V_ + r) * ND_ + kk]) : (unsigned short)0;
    sBt[i] = (kk < ND_) ? f2bf(nv2o[(bp * ND_ + kk) * V_ + r]) : (unsigned short)0;
  }
  __syncthreads();

  int wave = tid >> 5, lane = tid & 31, half = lane >> 4, l15 = lane & 15;
  for (int t = wave; t < 16; t += 8) {
    int mt = t >> 2, nt = t & 3;
    BfFrag a, b;
    const unsigned short* ar = sA + (mt * 16 + l15) * 32 + half * 8;
    const unsigned short* br = sBt + (nt * 16 + l15) * 32 + half * 8;
    a.q[0] = *(const uint4*)(ar);
    a.q[1] = *(const uint4*)(ar + 16);
    b.q[0] = *(const uint4*)(br);
    b.q[1] = *(const uint4*)(br + 16);
    v8f c = {0.f, 0.f, 0.f, 0.f, 0.f, 0.f, 0.f, 0.f};
    c = __builtin_amdgcn_wmma_f32_16x16x32_bf16(false, a.v, false, b.v,
                                                (short)0, c, false, false);
#pragma unroll
    for (int i = 0; i < 8; ++i)
      sS[(mt * 16 + half * 8 + i) * 66 + nt * 16 + l15] = fmaxf(c[i], 0.f);
  }
  __syncthreads();

  if (tid < 64) {
    int v = tid;
    float mx = -1e30f;
    for (int w = 0; w < 64; ++w) mx = fmaxf(mx, sS[v * 66 + w]);
    float sum = 0.f;
    for (int w = 0; w < 64; ++w) {
      float e = __expf(sS[v * 66 + w] - mx);
      sS[v * 66 + w] = e;
      sum += e;
    }
    float inv = 1.f / sum;
    for (int w = 0; w < 64; ++w)
      adpT[(bp * V_ + w) * V_ + v] = f2bf(sS[v * 66 + w] * inv);
  }
}

// -------------------------------------------------------------------------
// GCN hop: xout[b,w,p,:] = sum_v adp[b,p,v,w] * xin[b,v,p,:]
// A = adpT[bp] (64x64, m=w,k=v, k-contiguous); B staged transposed [d][v]
// so B-frags are k-contiguous. grid = B*P, block = 256.
__global__ void hop_kernel(const unsigned short* __restrict__ xin,
                           const unsigned short* __restrict__ adpT,
                           unsigned short* __restrict__ xout) {
  __shared__ __align__(16) unsigned short sA[64 * 64];    // [w][v]
  __shared__ __align__(16) unsigned short sBt[128 * 64];  // [d][v]
  const int tid = threadIdx.x;
  size_t bp = blockIdx.x;
  int b = (int)(bp / P_), p = (int)(bp % P_);
  for (int i = tid; i < 512; i += 256)
    ((uint4*)sA)[i] = ((const uint4*)(adpT + bp * 4096))[i];
  for (int i = tid; i < 1024; i += 256) {
    int v = i >> 4, c = i & 15;  // 64 rows x 16 chunks of 8
    U4x8 xu;
    xu.q = *(const uint4*)(xin + (((size_t)b * V_ + v) * P_ + p) * D_ + c * 8);
#pragma unroll
    for (int e = 0; e < 8; ++e) sBt[(c * 8 + e) * 64 + v] = xu.u[e];
  }
  __syncthreads();

  int wave = tid >> 5, lane = tid & 31, half = lane >> 4, l15 = lane & 15;
  int mt = wave & 3;
  int ntb = (wave >> 2) * 4;
  for (int nt = ntb; nt < ntb + 4; ++nt) {
    v8f c = {0.f, 0.f, 0.f, 0.f, 0.f, 0.f, 0.f, 0.f};
    for (int kk = 0; kk < 64; kk += 32) {
      BfFrag a, b;
      const unsigned short* ar = sA + (mt * 16 + l15) * 64 + kk + half * 8;
      const unsigned short* br = sBt + (nt * 16 + l15) * 64 + kk + half * 8;
      a.q[0] = *(const uint4*)(ar);
      a.q[1] = *(const uint4*)(ar + 16);
      b.q[0] = *(const uint4*)(br);
      b.q[1] = *(const uint4*)(br + 16);
      c = __builtin_amdgcn_wmma_f32_16x16x32_bf16(false, a.v, false, b.v,
                                                  (short)0, c, false, false);
    }
#pragma unroll
    for (int i = 0; i < 8; ++i) {
      int w = mt * 16 + half * 8 + i;
      xout[(((size_t)b * V_ + w) * P_ + p) * D_ + nt * 16 + l15] = f2bf(c[i]);
    }
  }
}

// -------------------------------------------------------------------------
extern "C" void kernel_launch(void* const* d_in, const int* in_sizes, int n_in,
                              void* d_out, int out_size, void* d_ws, size_t ws_size,
                              hipStream_t stream) {
  (void)in_sizes; (void)n_in; (void)out_size; (void)ws_size;
  const float* x      = (const float*)d_in[0];
  const float* nv1    = (const float*)d_in[1];
  const float* nv2    = (const float*)d_in[2];
  const float* Wq = (const float*)d_in[3];  const float* bq = (const float*)d_in[4];
  const float* Wk = (const float*)d_in[5];  const float* bk = (const float*)d_in[6];
  const float* Wv = (const float*)d_in[7];  const float* bvp = (const float*)d_in[8];
  const float* Wo = (const float*)d_in[9];  const float* bo = (const float*)d_in[10];
  const float* ln1g = (const float*)d_in[11]; const float* ln1b = (const float*)d_in[12];
  const float* Wff1 = (const float*)d_in[13]; const float* bff1 = (const float*)d_in[14];
  const float* Wff2 = (const float*)d_in[15]; const float* bff2 = (const float*)d_in[16];
  const float* ln2g = (const float*)d_in[17]; const float* ln2b = (const float*)d_in[18];
  const float* g1W = (const float*)d_in[19]; const float* g1b = (const float*)d_in[20];
  const float* g2W = (const float*)d_in[21]; const float* g2b = (const float*)d_in[22];
  const float* lin1W = (const float*)d_in[23]; const float* lin1b = (const float*)d_in[24];
  const float* lin2W = (const float*)d_in[25]; const float* lin2b = (const float*)d_in[26];
  const float* gcnW = (const float*)d_in[27]; const float* gcnb = (const float*)d_in[28];
  const int* patch = (const int*)d_in[29];
  const int* stay  = (const int*)d_in[30];
  float* out = (float*)d_out;

  const size_t BVPD = (size_t)B_ * V_ * P_ * D_;  // 16,777,216
  const size_t M    = (size_t)B_ * V_ * P_;       // 131,072

  // ---- workspace carve ----
  char* wsp = (char*)d_ws;
  auto carve = [&](size_t bytes) -> void* {
    void* p = (void*)wsp;
    wsp += (bytes + 255) & ~(size_t)255;
    return p;
  };
  float*          X    = (float*)carve(BVPD * 4);           // h -> h1 -> xe
  unsigned short* Xbf  = (unsigned short*)carve(BVPD * 2);
  unsigned short* Qbf  = (unsigned short*)carve(BVPD * 2);  // later: x1 bf16
  unsigned short* Kbf  = (unsigned short*)carve(BVPD * 2);  // later: x2 bf16
  unsigned short* Vbf  = (unsigned short*)carve(BVPD * 2);
  unsigned short* Obf  = (unsigned short*)carve(BVPD * 2);
  float*          T    = (float*)carve(BVPD * 4);           // fp32 tmp / Tg
  unsigned short* FFbf = (unsigned short*)carve(M * FF_ * 2);
  unsigned short* AdpT = (unsigned short*)carve((size_t)B_ * P_ * V_ * V_ * 2);
  float*          Nv1o = (float*)carve((size_t)B_ * P_ * V_ * ND_ * 4);
  float*          Nv2o = (float*)carve((size_t)B_ * P_ * V_ * ND_ * 4);
  float*          Gvc1 = (float*)carve(V_ * 4);
  float*          Gvc2 = (float*)carve(V_ * 4);
  unsigned short* Wqb  = (unsigned short*)carve(D_ * D_ * 2);
  unsigned short* Wkb  = (unsigned short*)carve(D_ * D_ * 2);
  unsigned short* Wvb  = (unsigned short*)carve(D_ * D_ * 2);
  unsigned short* Wob  = (unsigned short*)carve(D_ * D_ * 2);
  unsigned short* Wf1b = (unsigned short*)carve(D_ * FF_ * 2);
  unsigned short* Wf2b = (unsigned short*)carve(FF_ * D_ * 2);
  unsigned short* Wgb  = (unsigned short*)carve(3 * D_ * D_ * 2);
  unsigned short* Wpk  = (unsigned short*)carve(D_ * 64 * 2);
  float*          Bpk  = (float*)carve(64 * 4);

  // ---- weight conversion / packing ----
  auto cvt = [&](const float* s, unsigned short* d, int n) {
    cvt_bf16_kernel<<<dim3((n + 255) / 256), dim3(256), 0, stream>>>(s, d, n);
  };
  cvt(Wq, Wqb, D_ * D_);
  cvt(Wk, Wkb, D_ * D_);
  cvt(Wv, Wvb, D_ * D_);
  cvt(Wo, Wob, D_ * D_);
  cvt(Wff1, Wf1b, D_ * FF_);
  cvt(Wff2, Wf2b, FF_ * D_);
  cvt(gcnW, Wgb, 3 * D_ * D_);
  packgate_kernel<<<dim3((D_ * 64 + 255) / 256), dim3(256), 0, stream>>>(
      lin1W, lin2W, g1W, g2W, lin1b, lin2b, g1b, g2b, Wpk, Bpk);

  // ---- h = x + PE ----
  pe_kernel<<<dim3((unsigned)M), dim3(D_), 0, stream>>>(x, X, Xbf);

  const dim3 blk(256);
  const dim3 gD(M / 128, D_ / 32);
  const dim3 gFF(M / 128, FF_ / 32);
  const dim3 gG(M / 128, 2);                 // N=64 gating GEMM
  const size_t smD = (size_t)D_ * 32 * 2;    // K=128 panel
  const size_t smFF = (size_t)FF_ * 32 * 2;  // K=512 panel

  // ---- Q, K, V projections (bf16 out) ----
  gemm_bf16_kernel<GF_BF16><<<gD, blk, smD, stream>>>(Xbf, Wqb, bq, nullptr,
                                                      Qbf, (int)M, D_, D_);
  gemm_bf16_kernel<GF_BF16><<<gD, blk, smD, stream>>>(Xbf, Wkb, bk, nullptr,
                                                      Kbf, (int)M, D_, D_);
  gemm_bf16_kernel<GF_BF16><<<gD, blk, smD, stream>>>(Xbf, Wvb, bvp, nullptr,
                                                      Vbf, (int)M, D_, D_);

  // ---- attention ----
  attention_kernel<<<dim3(B_ * V_, H_, P_ / 128), blk, 0, stream>>>(
      Qbf, Kbf, Vbf, Obf, patch, stay);

  // ---- o @ Wo -> T ; h1 = LN(h + T) ----
  gemm_bf16_kernel<GF_F32><<<gD, blk, smD, stream>>>(Obf, Wob, bo, T, nullptr,
                                                     (int)M, D_, D_);
  ln_kernel<<<dim3((unsigned)M), dim3(D_), 0, stream>>>(X, T, ln1g, ln1b, X, Xbf);

  // ---- FF ----
  gemm_bf16_kernel<GF_RELU | GF_BF16><<<gFF, blk, smD, stream>>>(
      Xbf, Wf1b, bff1, nullptr, FFbf, (int)M, FF_, D_);
  gemm_bf16_kernel<GF_F32><<<gD, blk, smFF, stream>>>(FFbf, Wf2b, bff2, T,
                                                      nullptr, (int)M, D_, FF_);
  ln_kernel<<<dim3((unsigned)M), dim3(D_), 0, stream>>>(X, T, ln2g, ln2b, X, Xbf);
  // X/Xbf now hold xe

  // ---- gated nodevec update as one WMMA GEMM + cheap finish ----
  gateconst_kernel<<<dim3(1), dim3(64), 0, stream>>>(nv1, nv2, g1W, g2W, Gvc1, Gvc2);
  gemm_bf16_kernel<GF_F32><<<gG, blk, smD, stream>>>(Xbf, Wpk, Bpk, T, nullptr,
                                                     (int)M, 64, D_);
  gatefin_kernel<<<dim3((unsigned)(M * 16 / 256)), blk, 0, stream>>>(
      T, nv1, nv2, Gvc1, Gvc2, Nv1o, Nv2o);
  adj_kernel<<<dim3(B_ * P_), blk, 0, stream>>>(Nv1o, Nv2o, AdpT);

  // ---- 2-hop GCN (reuse Qbf/Kbf as x1/x2) ----
  unsigned short* X1bf = Qbf;
  unsigned short* X2bf = Kbf;
  hop_kernel<<<dim3(B_ * P_), blk, 0, stream>>>(Xbf, AdpT, X1bf);
  hop_kernel<<<dim3(B_ * P_), blk, 0, stream>>>(X1bf, AdpT, X2bf);

  // ---- out = xe@W0 + x1@W1 + x2@W2 + b  (gcn_W split along rows) ----
  gemm_bf16_kernel<GF_F32><<<gD, blk, smD, stream>>>(Xbf, Wgb, gcnb, out,
                                                     nullptr, (int)M, D_, D_);
  gemm_bf16_kernel<GF_F32 | GF_ACC><<<gD, blk, smD, stream>>>(
      X1bf, Wgb + D_ * D_, nullptr, out, nullptr, (int)M, D_, D_);
  gemm_bf16_kernel<GF_F32 | GF_ACC><<<gD, blk, smD, stream>>>(
      X2bf, Wgb + 2 * D_ * D_, nullptr, out, nullptr, (int)M, D_, D_);
}
